// EnergyBasedConfidence_9431748182489
// MI455X (gfx1250) — matrix operand order repaired and checked by reading, not compiled, not run on hardware
//
#include <hip/hip_runtime.h>

#define H_DIM  1024
#define HD     512
#define NTF    11
#define NPAIR  55
#define NB     8            // batches per workgroup
#define ROWS   (NB * NTF)   // 88 valid M rows
#define MT     6            // 6 M-tiles of 16 -> 96 padded rows
#define THREADS 128         // 4 wave32

// ws layout (ushort units):
//   wsKt [512][1024]   @ 0
//   wsLt [512][1024]   @ 512*1024
//   wsAt [1024][1024]  @ 2*512*1024
//   wsBt [1024][1024]  @ 2*512*1024 + 1024*1024
//   zrow [1024]        @ WS_ZROW
//   hb   [B*11][1024]  @ WS_HB
#define WS_ZROW ((size_t)2 * HD * H_DIM + (size_t)2 * H_DIM * H_DIM)
#define WS_HB   (WS_ZROW + H_DIM)

typedef __attribute__((ext_vector_type(16))) __bf16 v16bf;
typedef __attribute__((ext_vector_type(8)))  float  v8f;

__device__ __forceinline__ unsigned int f2bf(float f) {
  union { float f; unsigned int u; } c; c.f = f;
  return (c.u + 0x7FFFu + ((c.u >> 16) & 1u)) >> 16;   // RNE f32 -> bf16
}

union BFrag { v16bf v; uint4 q[2]; };

// ---------------------------------------------------------------------------
// Kernel 0: weights -> bf16 transposed [n][k]; also zero-fills the pad row.
// ---------------------------------------------------------------------------
__global__ void wcvt_kernel(const float* __restrict__ kW1,
                            const float* __restrict__ lW1,
                            const float* __restrict__ dW1,
                            unsigned short* __restrict__ ws) {
  unsigned short* wsKt = ws;
  unsigned short* wsLt = ws + (size_t)HD * H_DIM;
  unsigned short* wsAt = ws + (size_t)2 * HD * H_DIM;
  unsigned short* wsBt = wsAt + (size_t)H_DIM * H_DIM;
  int idx = blockIdx.x * blockDim.x + threadIdx.x;
  if (idx >= H_DIM * H_DIM) return;
  int n = idx >> 10;
  int k = idx & (H_DIM - 1);
  wsAt[idx] = (unsigned short)f2bf(dW1[(size_t)k * H_DIM + n]);            // dW1a
  wsBt[idx] = (unsigned short)f2bf(dW1[(size_t)(k + H_DIM) * H_DIM + n]);  // dW1b
  if (n < HD) {
    wsKt[idx] = (unsigned short)f2bf(kW1[(size_t)k * HD + n]);
    wsLt[idx] = (unsigned short)f2bf(lW1[(size_t)k * HD + n]);
  }
  if (idx < H_DIM) ws[WS_ZROW + idx] = 0;   // zero pad-row for padded M rows
}

// ---------------------------------------------------------------------------
// Kernel 1: h_stack f32 -> bf16 rows (8 elements per thread, b128 in/out)
// ---------------------------------------------------------------------------
__global__ void hcvt_kernel(const float* __restrict__ h,
                            unsigned short* __restrict__ hb, int total8) {
  int idx = blockIdx.x * blockDim.x + threadIdx.x;
  if (idx >= total8) return;
  size_t base = (size_t)idx * 8;
  float4 f0 = *(const float4*)(h + base);
  float4 f1 = *(const float4*)(h + base + 4);
  uint4 o;
  o.x = f2bf(f0.x) | (f2bf(f0.y) << 16);
  o.y = f2bf(f0.z) | (f2bf(f0.w) << 16);
  o.z = f2bf(f1.x) | (f2bf(f1.y) << 16);
  o.w = f2bf(f1.z) | (f2bf(f1.w) << 16);
  *(uint4*)(hb + base) = o;
}

// A fragment (16-bit A 16x32): lane<16 K {k0..k0+7}U{k0+16..k0+23}; caller adds
// +8 for hi half. Native bf16 rows -> two contiguous b128 loads.
__device__ __forceinline__ v16bf load_a_frag(const unsigned short* rp, int k0) {
  BFrag a;
  a.q[0] = *(const uint4*)(rp + k0);
  a.q[1] = *(const uint4*)(rp + k0 + 16);
  return a.v;
}

// B fragment (16-bit B 32x16): lane<16 K {k0..k0+7}U{k0+16..k0+23}; caller adds
// +8 for hi half. Transposed [n][k] rows -> two contiguous b128 loads.
__device__ __forceinline__ v16bf load_b_frag(const unsigned short* wp, int k0) {
  BFrag b;
  b.q[0] = *(const uint4*)(wp + k0);
  b.q[1] = *(const uint4*)(wp + k0 + 16);
  return b.v;
}

#define WMMA_BF16(A, B, C) \
  __builtin_amdgcn_wmma_f32_16x16x32_bf16(false, (A), false, (B), (short)0, (C), false, false)

// ---------------------------------------------------------------------------
// Kernel 2: fully fused energy pipeline. One WG = 8 batches (88 M rows).
// Each wave owns two 16-col subtiles, 64 columns apart.
// ---------------------------------------------------------------------------
__global__ __launch_bounds__(THREADS) void fused_energy_kernel(
    const unsigned short* __restrict__ hb, const unsigned short* __restrict__ zrow,
    const float* __restrict__ bconf,
    const float* __restrict__ kb1, const float* __restrict__ kW2, const float* __restrict__ kb2,
    const float* __restrict__ lb1, const float* __restrict__ lW2, const float* __restrict__ lb2,
    const float* __restrict__ db1, const float* __restrict__ dW2, const float* __restrict__ db2,
    const float* __restrict__ kwp, const float* __restrict__ lwp,
    const float* __restrict__ iwp, const float* __restrict__ tpp,
    const unsigned short* __restrict__ wsKt, const unsigned short* __restrict__ wsLt,
    const unsigned short* __restrict__ wsAt, const unsigned short* __restrict__ wsBt,
    float* __restrict__ out, int Btot)
{
  __shared__ float lds_pi[MT * 16 * 64];   // 24 KB
  __shared__ float lds_pj[MT * 16 * 64];   // 24 KB
  __shared__ float lds_db1[64];
  __shared__ float lds_dw2[64];
  __shared__ float rowacc_k[MT * 16];
  __shared__ float rowacc_l[MT * 16];
  __shared__ float pairacc[NB * NPAIR];

  const int tid  = threadIdx.x;
  const int lane = tid & 31;
  const int wv   = tid >> 5;            // wave 0..3
  const int lr   = lane & 15;
  const int hi   = lane >> 4;           // 0 or 1
  const int ka   = hi << 3;             // A-fragment k sub-offset (0/8)
  const int kbo  = hi << 3;             // B-fragment k sub-offset (0/8)
  const int b0   = blockIdx.x * NB;

  for (int i = tid; i < MT * 16; i += THREADS) { rowacc_k[i] = 0.f; rowacc_l[i] = 0.f; }
  for (int i = tid; i < NB * NPAIR; i += THREADS) pairacc[i] = 0.f;
  __syncthreads();

  // A-row base pointers (M = lane%16 within each tile); padded rows -> zero row
  const unsigned short* arow[MT];
#pragma unroll
  for (int m = 0; m < MT; ++m) {
    int rl = m * 16 + lr;
    arow[m] = (rl < ROWS) ? (hb + ((size_t)(b0 * NTF + rl)) * H_DIM) : zrow;
  }

  // ---------------- Phase A: kinetic + local hidden layers (N = 512) -------
#pragma unroll 1
  for (int nt = 0; nt < HD / 128; ++nt) {
    const int ncol0 = nt * 128 + wv * 16 + lr;   // subtile 0 column
    const unsigned short* wk0 = wsKt + (size_t)ncol0 * H_DIM;
    const unsigned short* wl0 = wsLt + (size_t)ncol0 * H_DIM;
    const unsigned short* wk1 = wk0 + (size_t)64 * H_DIM;
    const unsigned short* wl1 = wl0 + (size_t)64 * H_DIM;
    v8f accK[MT][2], accL[MT][2];
#pragma unroll
    for (int m = 0; m < MT; ++m)
#pragma unroll
      for (int e = 0; e < 8; ++e) {
        accK[m][0][e] = 0.f; accK[m][1][e] = 0.f;
        accL[m][0][e] = 0.f; accL[m][1][e] = 0.f;
      }

#pragma unroll 2
    for (int kk = 0; kk < H_DIM / 32; ++kk) {
      const int kb = kk * 32;
      v16bf bk0 = load_b_frag(wk0, kb + kbo);
      v16bf bk1 = load_b_frag(wk1, kb + kbo);
      v16bf bl0 = load_b_frag(wl0, kb + kbo);
      v16bf bl1 = load_b_frag(wl1, kb + kbo);
#pragma unroll
      for (int m = 0; m < MT; ++m) {
        v16bf a = load_a_frag(arow[m], kb + ka);
        accK[m][0] = WMMA_BF16(a, bk0, accK[m][0]);
        accK[m][1] = WMMA_BF16(a, bk1, accK[m][1]);
        accL[m][0] = WMMA_BF16(a, bl0, accL[m][0]);
        accL[m][1] = WMMA_BF16(a, bl1, accL[m][1]);
      }
    }
    // epilogue: bias + relu, fold 512->1 GEMV, reduce 16 cols via shuffles
#pragma unroll
    for (int t = 0; t < 2; ++t) {
      const int ncol = ncol0 + t * 64;
      const float kb1c = kb1[ncol], kw2c = kW2[ncol];
      const float lb1c = lb1[ncol], lw2c = lW2[ncol];
#pragma unroll
      for (int m = 0; m < MT; ++m) {
#pragma unroll
        for (int v = 0; v < 8; ++v) {
          int rl = m * 16 + v + (hi << 3);               // C layout: M = v (+8 hi-half)
          float hk = accK[m][t][v] + kb1c; hk = hk > 0.f ? hk * kw2c : 0.f;
          float hl = accL[m][t][v] + lb1c; hl = hl > 0.f ? hl * lw2c : 0.f;
          for (int s = 1; s < 16; s <<= 1) { hk += __shfl_xor(hk, s, 32); hl += __shfl_xor(hl, s, 32); }
          if (lr == 0) { atomicAdd(&rowacc_k[rl], hk); atomicAdd(&rowacc_l[rl], hl); }
        }
      }
    }
  }

  // ---------------- Phase B: pre_i / pre_j + pair interaction (N = 1024) ---
#pragma unroll 1
  for (int nt = 0; nt < H_DIM / 128; ++nt) {
    const int ncol0 = nt * 128 + wv * 16 + lr;
    const unsigned short* wi0 = wsAt + (size_t)ncol0 * H_DIM;
    const unsigned short* wj0 = wsBt + (size_t)ncol0 * H_DIM;
    const unsigned short* wi1 = wi0 + (size_t)64 * H_DIM;
    const unsigned short* wj1 = wj0 + (size_t)64 * H_DIM;
    v8f accI[MT][2], accJ[MT][2];
#pragma unroll
    for (int m = 0; m < MT; ++m)
#pragma unroll
      for (int e = 0; e < 8; ++e) {
        accI[m][0][e] = 0.f; accI[m][1][e] = 0.f;
        accJ[m][0][e] = 0.f; accJ[m][1][e] = 0.f;
      }

#pragma unroll 2
    for (int kk = 0; kk < H_DIM / 32; ++kk) {
      const int kb = kk * 32;
      v16bf bi0 = load_b_frag(wi0, kb + kbo);
      v16bf bi1 = load_b_frag(wi1, kb + kbo);
      v16bf bj0 = load_b_frag(wj0, kb + kbo);
      v16bf bj1 = load_b_frag(wj1, kb + kbo);
#pragma unroll
      for (int m = 0; m < MT; ++m) {
        v16bf a = load_a_frag(arow[m], kb + ka);
        accI[m][0] = WMMA_BF16(a, bi0, accI[m][0]);
        accI[m][1] = WMMA_BF16(a, bi1, accI[m][1]);
        accJ[m][0] = WMMA_BF16(a, bj0, accJ[m][0]);
        accJ[m][1] = WMMA_BF16(a, bj1, accJ[m][1]);
      }
    }
    // two 64-col groups: stage 96x64 tile of pre_i/pre_j, then pair-reduce
    const int cl = wv * 16 + lr;
#pragma unroll 1
    for (int t = 0; t < 2; ++t) {
      const int dbase = nt * 128 + t * 64;
      if (tid < 64) { lds_db1[tid] = db1[dbase + tid]; lds_dw2[tid] = dW2[dbase + tid]; }
#pragma unroll
      for (int m = 0; m < MT; ++m)
#pragma unroll
        for (int v = 0; v < 8; ++v) {
          int rl = m * 16 + v + (hi << 3);
          lds_pi[rl * 64 + cl] = accI[m][t][v];
          lds_pj[rl * 64 + cl] = accJ[m][t][v];
        }
      __syncthreads();

      // separable-over-d pair reduction: sum_d relu(pi+pj+db1[d]) * dW2[d]
      for (int item = tid; item < NB * NPAIR; item += THREADS) {
        int lb = item / NPAIR, p = item % NPAIR;
        int ti = 0, off = 0;
        while (p >= off + (NTF - 1 - ti)) { off += NTF - 1 - ti; ++ti; }
        int tj = ti + 1 + (p - off);
        const float* ri = &lds_pi[(lb * NTF + ti) * 64];
        const float* rj = &lds_pj[(lb * NTF + tj) * 64];
        float s = 0.f;
        for (int d = 0; d < 64; ++d) {
          float a = ri[d] + rj[d] + lds_db1[d];
          if (a > 0.f) s += a * lds_dw2[d];
        }
        pairacc[item] += s;   // same thread owns same item every pass: no race
      }
      __syncthreads();
    }
  }

  // ---------------- Final per-batch reduction ------------------------------
  if (tid < NB) {
    const int lb = tid, b = b0 + lb;
    const float kb2v = kb2[0], lb2v = lb2[0], db2v = db2[0];
    const float kwv = kwp[0], lwv = lwp[0], iwv = iwp[0], Tv = tpp[0];
    float ksum = 0.f, lsum = 0.f;
    for (int t = 0; t < NTF; ++t) {
      int rl = lb * NTF + t;
      ksum += rowacc_k[rl] + kb2v;
      float lp = rowacc_l[rl] + lb2v;
      lsum += lp > 0.f ? lp : 0.f;
    }
    float kin = -kwv * (ksum / (float)NTF);
    float loc =  lwv * (lsum / (float)NTF);
    float isum = 0.f;
    int p = 0;
    for (int ti = 0; ti < NTF - 1; ++ti)
      for (int tj = ti + 1; tj < NTF; ++tj, ++p) {
        float dis = pairacc[lb * NPAIR + p] + db2v;
        dis = dis > 0.f ? dis : 0.f;
        isum += dis * (1.0f / (float)(tj - ti + 1));
      }
    float inter = iwv * (isum / (float)NPAIR);
    float tot = kin + loc + inter;
    float ec = 1.f / (1.f + expf(tot / (fabsf(Tv) + 0.1f)));
    out[0 * Btot + b] = tot;
    out[1 * Btot + b] = kin;
    out[2 * Btot + b] = loc;
    out[3 * Btot + b] = inter;
    out[4 * Btot + b] = ec;
    out[5 * Btot + b] = bconf[b] * ec;
  }
}

extern "C" void kernel_launch(void* const* d_in, const int* in_sizes, int n_in,
                              void* d_out, int out_size, void* d_ws, size_t ws_size,
                              hipStream_t stream) {
  (void)n_in; (void)out_size; (void)ws_size;
  const float* h    = (const float*)d_in[0];
  const float* bcnf = (const float*)d_in[1];
  const float* kW1  = (const float*)d_in[2];
  const float* kb1  = (const float*)d_in[3];
  const float* kW2  = (const float*)d_in[4];
  const float* kb2  = (const float*)d_in[5];
  const float* lW1  = (const float*)d_in[6];
  const float* lb1  = (const float*)d_in[7];
  const float* lW2  = (const float*)d_in[8];
  const float* lb2  = (const float*)d_in[9];
  const float* dW1  = (const float*)d_in[10];
  const float* db1  = (const float*)d_in[11];
  const float* dW2  = (const float*)d_in[12];
  const float* db2  = (const float*)d_in[13];
  const float* kw   = (const float*)d_in[14];
  const float* lw   = (const float*)d_in[15];
  const float* iw   = (const float*)d_in[16];
  const float* tp   = (const float*)d_in[17];
  unsigned short* ws = (unsigned short*)d_ws;
  const int Btot = in_sizes[1];

  // Weights -> bf16 transposed (6 MB) + zero pad-row
  int cvtThreads = 256;
  int cvtBlocks = (H_DIM * H_DIM + cvtThreads - 1) / cvtThreads;
  wcvt_kernel<<<cvtBlocks, cvtThreads, 0, stream>>>(kW1, lW1, dW1, ws);

  // h_stack -> bf16 (44 MB)
  unsigned short* hb = ws + WS_HB;
  int total8 = (Btot * NTF * H_DIM) / 8;
  int hBlocks = (total8 + cvtThreads - 1) / cvtThreads;
  hcvt_kernel<<<hBlocks, cvtThreads, 0, stream>>>(h, hb, total8);

  const unsigned short* wsKt = ws;
  const unsigned short* wsLt = ws + (size_t)HD * H_DIM;
  const unsigned short* wsAt = ws + (size_t)2 * HD * H_DIM;
  const unsigned short* wsBt = wsAt + (size_t)H_DIM * H_DIM;
  const unsigned short* zrow = ws + WS_ZROW;

  int nblk = Btot / NB;
  fused_energy_kernel<<<nblk, THREADS, 0, stream>>>(
      hb, zrow, bcnf, kb1, kW2, kb2, lb1, lW2, lb2, db1, dW2, db2,
      kw, lw, iw, tp, wsKt, wsLt, wsAt, wsBt, (float*)d_out, Btot);
}
